// FillPolygon_11922829213956
// MI455X (gfx1250) — matrix operand order repaired and checked by reading, not compiled, not run on hardware
//
#include <hip/hip_runtime.h>
#include <hip/hip_bf16.h>
#include <math.h>

typedef __attribute__((ext_vector_type(16))) _Float16 v16h;
typedef __attribute__((ext_vector_type(8)))  float    v8f;

#define BATCH   2
#define HDIM    256
#define WDIM    256
#define NVERT   256
#define K3      (3 * NVERT)            // 768 extended vertex samples
#define TWO_PI_F 6.28318530717958647692f
#define HALF_F   128.0f
#define NEG100_OVER_LN2 (-144.26950408889634074f)   // -100 / ln(2)

// One block = 256 threads = 8 waves; each wave handles 16 pixels.
// Blocks per batch = 65536 / 128 = 512; total grid = 1024 blocks.
__global__ __launch_bounds__(256)
void FillPolygon_11922829213956_kernel(const float* __restrict__ in,  // (B, V, 2)
                                       float* __restrict__ out) {    // (B, 1, H, W)
    __shared__ float                  s_tt3[K3];   // extended angles (f32)
    __shared__ __align__(32) _Float16 s_r3h[K3];   // extended radii (f16, B-matrix)

    const int tid = threadIdx.x;
    const int b   = blockIdx.x >> 9;               // 512 blocks per batch

    // ---- Stage vertex tables into LDS (one vertex per thread) ----
    {
        const int v = tid;                         // V == blockDim.x == 256
        float y = in[(b * NVERT + v) * 2 + 0];
        float x = in[(b * NVERT + v) * 2 + 1];
        float xv = x - HALF_F;                     // x - width/2
        float yv = y - HALF_F;                     // y - height/2
        float r  = sqrtf(xv * xv + yv * yv);
        float tt = atan2f(yv, xv);
        tt = (tt < 0.0f) ? tt + TWO_PI_F : tt;     // _fix_radians -> [0, 2pi)
        s_tt3[v]             = tt - TWO_PI_F;
        s_tt3[NVERT + v]     = tt;
        s_tt3[2 * NVERT + v] = tt + TWO_PI_F;
        _Float16 rh = (_Float16)r;
        s_r3h[v]             = rh;
        s_r3h[NVERT + v]     = rh;
        s_r3h[2 * NVERT + v] = rh;
    }
    __syncthreads();

    // ---- Per-wave pixel tile ----
    const int wave  = tid >> 5;                    // 0..7
    const int lane  = tid & 31;
    const int laneM = lane & 15;
    const bool hi   = lane >= 16;

    const int tileInB = ((blockIdx.x & 511) << 3) + wave;  // 0..4095
    const int p       = (tileInB << 4) + laneM;            // pixel index in batch
    // torch/jax grid construction: y varies fastest -> y = p%W, x = p/W
    const float ym = (float)(p & (WDIM - 1)) - HALF_F;
    const float xm = (float)(p >> 8) - HALF_F;
    const float r_mat = sqrtf(xm * xm + ym * ym);
    float ttm = atan2f(ym, xm);
    ttm = (ttm < 0.0f) ? ttm + TWO_PI_F : ttm;

    v8f c = {};                                    // f32 acc: N0 = sum s*r, N1 = sum s

    const int kaOfs = hi ? 16 : 0;                 // A-matrix lane K offset
    const int kbOfs = hi ? 16 : 0;                 // B-matrix lane K offset

    // Branchless column masks for B: col 0 <- r3, col 1 <- 1.0, rest 0.
    const _Float16 m0 = (laneM == 0) ? (_Float16)1.0f : (_Float16)0.0f;
    const _Float16 m1 = (laneM == 1) ? (_Float16)1.0f : (_Float16)0.0f;

    #pragma unroll 4
    for (int kb = 0; kb < K3 / 32; ++kb) {
        const int kbase = kb * 32;

        // A: s = exp2(max(d^2 * (-100/ln2), -100/ln2)) == exp(-100*min(d^2,1)).
        // Raw v_exp_f32 (no OCML subnormal fixup): args in [-144.27, 0]; any
        // result < 2^-24 flushes to 0 in the f16 convert anyway.
        v16h a;
        #pragma unroll
        for (int h = 0; h < 16; ++h) {
            const int k = kbase + kaOfs + h;             // K = h + (hi?16:0)
            float d  = ttm - s_tt3[k];
            float y  = fmaxf(d * d * NEG100_OVER_LN2, NEG100_OVER_LN2);
            a[h] = (_Float16)__builtin_amdgcn_exp2f(y); // single v_exp_f32
        }

        // B: bm = r3_vec * m0 + m1   (pure v_pk_fma_f16, no EXEC games)
        const v16h rv = *(const v16h*)&s_r3h[kbase + kbOfs];  // 2x ds_load_b128
        const v16h bm = rv * m0 + m1;

        // D(16x16,f32) += A(16x32,f16) x B(32x16,f16) -- co-executes with exp stream
        c = __builtin_amdgcn_wmma_f32_16x16x32_f16(
                /*neg_a=*/false, a, /*neg_b=*/false, bm,
                /*c_mod=*/(short)0, c, /*reuse_a=*/false, /*reuse_b=*/false);
    }

    // ---- Extract per-pixel sums from C layout, branchlessly ----
    // C: VGPR i holds (M = i + (hi?8:0), N = laneM). Pixel m: ssr at N=0, ss at N=1.
    float ssr_sel = 0.0f;
    float ss_sel  = 1.0f;
    #pragma unroll
    for (int m = 0; m < 16; ++m) {
        float ssr = __shfl(c[m & 7], (m < 8) ? 0 : 16, 32);
        float ss  = __shfl(c[m & 7], (m < 8) ? 1 : 17, 32);
        ssr_sel = (lane == m) ? ssr : ssr_sel;     // v_cndmask, no branches
        ss_sel  = (lane == m) ? ss  : ss_sel;
    }
    const float r_new = ssr_sel / fmaxf(ss_sel, 1e-20f);
    const float flag  = fminf(fmaxf(1.0f + r_new - r_mat, 0.0f), 1.0f);
    if (!hi) {
        out[b * (HDIM * WDIM) + p] = flag;         // lanes 0..15 own pixels 0..15
    }
}

extern "C" void kernel_launch(void* const* d_in, const int* in_sizes, int n_in,
                              void* d_out, int out_size, void* d_ws, size_t ws_size,
                              hipStream_t stream) {
    (void)in_sizes; (void)n_in; (void)d_ws; (void)ws_size; (void)out_size;
    const float* inputs = (const float*)d_in[0];   // (B, V, 2) float32
    float* out = (float*)d_out;                    // (B, 1, H, W) float32

    const int blocksPerBatch = (HDIM * WDIM) / 128; // 512
    dim3 grid(BATCH * blocksPerBatch);              // 1024
    dim3 block(256);                                // 8 waves, wave32
    FillPolygon_11922829213956_kernel<<<grid, block, 0, stream>>>(inputs, out);
}
